// TinyViTLayer_63814624084645
// MI455X (gfx1250) — compile-verified, matrix operand order checked by reading, not run on hardware
//
#include <hip/hip_runtime.h>
#include <hip/hip_bf16.h>
#include <hip/hip_fp16.h>

typedef __attribute__((ext_vector_type(16))) _Float16 v16h;
typedef __attribute__((ext_vector_type(8)))  float    v8f;

// Problem dims
constexpr int kB   = 64;
constexpr int kHW  = 784;        // 28*28
constexpr int kC   = 576;
constexpr int kNH  = 18;
constexpr int kN   = 49;         // window tokens
constexpr int kNP  = 64;         // padded window tokens
constexpr int kNW  = 1024;       // total windows (64 * 4 * 4)
constexpr int kHID = 2304;
constexpr int kTOK = kB * kHW;   // 50176
constexpr int kQKVC = 3 * kC;    // 1728

// ---------------------------------------------------------------------------
// WMMA fragment loader, 16-bit A/B operand layout per CDNA5 ISA 7.12.2:
// lane l (lr = l&15, hi = l>>4): elements 0..7 -> K = hi*8 + e
//                                elements 8..15 -> K = 16 + hi*8 + e
// Two contiguous 16-byte groups -> global_load_b128 pairs.
// ---------------------------------------------------------------------------
__device__ __forceinline__ v16h ldfrag(const _Float16* rowp, int hi) {
  const _Float16* p = rowp + (hi << 3);
  v16h r;
#pragma unroll
  for (int e = 0; e < 8; ++e) r[e] = p[e];
#pragma unroll
  for (int e = 0; e < 8; ++e) r[8 + e] = p[16 + e];
  return r;
}

__device__ __forceinline__ v8f wmma_f16(v16h a, v16h b, v8f c) {
  return __builtin_amdgcn_wmma_f32_16x16x32_f16(false, a, false, b, (short)0, c,
                                                false, false);
}

__device__ __forceinline__ float gelu_exact(float x) {
  return 0.5f * x * (1.0f + erff(x * 0.70710678118654752f));
}

// ---------------------------------------------------------------------------
// Generic WMMA GEMM: D[M,Nn] = A[M,K] * W[Nn,K]^T + bias, epilogues:
//  0: store f16   1: store f16 with exact GELU   2: store f32 + residual
// One wave computes a 32(M) x 64(N) tile: 2 A frags x 4 B frags -> 8 WMMA
// per K-step (6 fragment loads per 8 WMMAs).  M % 32 == 0, Nn % 64 == 0.
// ---------------------------------------------------------------------------
constexpr int EPI_F16 = 0, EPI_GELU = 1, EPI_RESF32 = 2;

template <int EPI>
__global__ __launch_bounds__(256) void gemm_wmma(
    const _Float16* __restrict__ A, const _Float16* __restrict__ W,
    const float* __restrict__ bias, void* __restrict__ outv,
    const float* __restrict__ res, int M, int K, int Nn) {
  int wave = (blockIdx.x * blockDim.x + threadIdx.x) >> 5;
  int lane = threadIdx.x & 31;
  int nStrips = Nn >> 6;
  int mt = wave / nStrips;
  int ns = wave - mt * nStrips;
  int m0 = mt << 5, n0 = ns << 6;
  if (m0 >= M) return;
  int lr = lane & 15, hi = lane >> 4;

  v8f acc[2][4] = {};
  const _Float16* arow0 = A + (size_t)(m0 + lr) * K;
  const _Float16* arow1 = arow0 + (size_t)16 * K;
  const _Float16* w0 = W + (size_t)(n0 + lr) * K;
  const _Float16* w1 = w0 + (size_t)16 * K;
  const _Float16* w2 = w0 + (size_t)32 * K;
  const _Float16* w3 = w0 + (size_t)48 * K;
  for (int k0 = 0; k0 < K; k0 += 32) {
    __builtin_prefetch(arow0 + k0 + 128, 0, 0);  // gfx1250 global_prefetch_b8
    __builtin_prefetch(arow1 + k0 + 128, 0, 0);
    v16h a0 = ldfrag(arow0 + k0, hi);
    v16h a1 = ldfrag(arow1 + k0, hi);
    v16h b0 = ldfrag(w0 + k0, hi);
    v16h b1 = ldfrag(w1 + k0, hi);
    v16h b2 = ldfrag(w2 + k0, hi);
    v16h b3 = ldfrag(w3 + k0, hi);
    acc[0][0] = wmma_f16(a0, b0, acc[0][0]);
    acc[0][1] = wmma_f16(a0, b1, acc[0][1]);
    acc[0][2] = wmma_f16(a0, b2, acc[0][2]);
    acc[0][3] = wmma_f16(a0, b3, acc[0][3]);
    acc[1][0] = wmma_f16(a1, b0, acc[1][0]);
    acc[1][1] = wmma_f16(a1, b1, acc[1][1]);
    acc[1][2] = wmma_f16(a1, b2, acc[1][2]);
    acc[1][3] = wmma_f16(a1, b3, acc[1][3]);
  }
  // D layout: lane holds col = n0 + t*16 + lr, VGPR i holds row base + hi*8 + i
  int cb = n0 + lr;
  float bi0 = bias[cb], bi1 = bias[cb + 16];
  float bi2 = bias[cb + 32], bi3 = bias[cb + 48];
#pragma unroll
  for (int mi = 0; mi < 2; ++mi) {
    int row0 = m0 + (mi << 4) + (hi << 3);
    if (EPI == EPI_RESF32) {
      float* out = (float*)outv;
#pragma unroll
      for (int i = 0; i < 8; ++i) {
        size_t r = (size_t)(row0 + i) * Nn;
        out[r + cb]      = acc[mi][0][i] + bi0 + res[r + cb];
        out[r + cb + 16] = acc[mi][1][i] + bi1 + res[r + cb + 16];
        out[r + cb + 32] = acc[mi][2][i] + bi2 + res[r + cb + 32];
        out[r + cb + 48] = acc[mi][3][i] + bi3 + res[r + cb + 48];
      }
    } else {
      _Float16* out = (_Float16*)outv;
#pragma unroll
      for (int i = 0; i < 8; ++i) {
        size_t r = (size_t)(row0 + i) * Nn;
        float v0 = acc[mi][0][i] + bi0, v1 = acc[mi][1][i] + bi1;
        float v2 = acc[mi][2][i] + bi2, v3 = acc[mi][3][i] + bi3;
        if (EPI == EPI_GELU) {
          v0 = gelu_exact(v0); v1 = gelu_exact(v1);
          v2 = gelu_exact(v2); v3 = gelu_exact(v3);
        }
        out[r + cb]      = (_Float16)v0;
        out[r + cb + 16] = (_Float16)v1;
        out[r + cb + 32] = (_Float16)v2;
        out[r + cb + 48] = (_Float16)v3;
      }
    }
  }
}

// ---------------------------------------------------------------------------
// Windowed LayerNorm: x[B,HW,C] -> h16[NW][64][C] (rows 49..63 zero-padded)
// ---------------------------------------------------------------------------
__global__ __launch_bounds__(256) void ln_win_kernel(
    const float* __restrict__ x, const float* __restrict__ g,
    const float* __restrict__ be, _Float16* __restrict__ h) {
  __shared__ float rs[8], rq[8];
  int w = blockIdx.x >> 6;
  int n = blockIdx.x & 63;
  int tid = threadIdx.x;
  _Float16* hrow = h + (size_t)blockIdx.x * kC;
  if (n >= kN) {
    for (int i = tid; i < kC; i += 256) hrow[i] = (_Float16)0.0f;
    return;
  }
  int b = w >> 4, wi = w & 15;
  int wh = wi >> 2, ww = wi & 3;
  int r = n / 7, cc = n - r * 7;
  int tokrow = (wh * 7 + r) * 28 + (ww * 7 + cc);
  const float* xr = x + ((size_t)b * kHW + tokrow) * kC;
  float s = 0.f, s2 = 0.f;
  for (int i = tid; i < kC; i += 256) { float v = xr[i]; s += v; s2 += v * v; }
  for (int off = 16; off >= 1; off >>= 1) {
    s  += __shfl_xor(s, off, 32);
    s2 += __shfl_xor(s2, off, 32);
  }
  if ((tid & 31) == 0) { rs[tid >> 5] = s; rq[tid >> 5] = s2; }
  __syncthreads();
  s = 0.f; s2 = 0.f;
#pragma unroll
  for (int i = 0; i < 8; ++i) { s += rs[i]; s2 += rq[i]; }
  float mean = s * (1.0f / kC);
  float var  = s2 * (1.0f / kC) - mean * mean;
  float inv  = rsqrtf(var + 1e-5f);
  for (int i = tid; i < kC; i += 256)
    hrow[i] = (_Float16)((xr[i] - mean) * inv * g[i] + be[i]);
}

// Token LayerNorm: x[T,C] -> h16[T,C]
__global__ __launch_bounds__(256) void ln_tok_kernel(
    const float* __restrict__ x, const float* __restrict__ g,
    const float* __restrict__ be, _Float16* __restrict__ h) {
  __shared__ float rs[8], rq[8];
  int tid = threadIdx.x;
  const float* xr = x + (size_t)blockIdx.x * kC;
  _Float16* hrow = h + (size_t)blockIdx.x * kC;
  float s = 0.f, s2 = 0.f;
  for (int i = tid; i < kC; i += 256) { float v = xr[i]; s += v; s2 += v * v; }
  for (int off = 16; off >= 1; off >>= 1) {
    s  += __shfl_xor(s, off, 32);
    s2 += __shfl_xor(s2, off, 32);
  }
  if ((tid & 31) == 0) { rs[tid >> 5] = s; rq[tid >> 5] = s2; }
  __syncthreads();
  s = 0.f; s2 = 0.f;
#pragma unroll
  for (int i = 0; i < 8; ++i) { s += rs[i]; s2 += rq[i]; }
  float mean = s * (1.0f / kC);
  float var  = s2 * (1.0f / kC) - mean * mean;
  float inv  = rsqrtf(var + 1e-5f);
  for (int i = tid; i < kC; i += 256)
    hrow[i] = (_Float16)((xr[i] - mean) * inv * g[i] + be[i]);
}

// ---------------------------------------------------------------------------
// Expand relative-position biases to a dense, pre-masked [NH][64][64] table:
//   col >= 49 -> -1e30 (softmax pad mask), row >= 49 -> 0, else gather.
// ---------------------------------------------------------------------------
__global__ __launch_bounds__(256) void expand_bias_kernel(
    const float* __restrict__ biases, const int* __restrict__ bidx,
    float* __restrict__ bf, int n_off) {
  int idx = blockIdx.x * 256 + threadIdx.x;
  if (idx >= kNH * kNP * kNP) return;
  int head = idx >> 12;
  int rc = idx & 4095;
  int row = rc >> 6, col = rc & 63;
  float v;
  if (col >= kN)      v = -1e30f;
  else if (row >= kN) v = 0.0f;
  else                v = biases[head * n_off + bidx[row * kN + col]];
  bf[idx] = v;
}

// ---------------------------------------------------------------------------
// Attention: one block (4 waves) per (window, head).
// V tile staged global->LDS with CDNA5 async-LDS loads (ASYNCcnt path),
// scores via WMMA (K=32 = head dim), dense pre-masked bias add, softmax
// in-register (wave32 half-group shuffles), P through LDS (layout transpose),
// attn@V via WMMA (K=64, 2 steps).
// ---------------------------------------------------------------------------
__global__ __launch_bounds__(128) void attn_kernel(
    const _Float16* __restrict__ qkv, const float* __restrict__ bias_full,
    _Float16* __restrict__ outp) {
  __shared__ _Float16 p_lds[64 * 64];
  __shared__ _Float16 v_lds[64 * 32];
  int w = blockIdx.x / kNH;
  int head = blockIdx.x - w * kNH;
  int tid = threadIdx.x;
  int wv = tid >> 5, lane = tid & 31;
  int lr = lane & 15, hi = lane >> 4;

  const _Float16* qb = qkv + (size_t)w * kNP * kQKVC + head * 96;
  const _Float16* kb = qb + 32;
  const _Float16* vb = qb + 64;

  // V tile (64 rows x 32 cols) -> LDS via async-LDS B128 loads (no VGPR
  // round-trip; INST_OFFSET applies to both LDS and global addresses).
  {
    int row = tid >> 1, half = (tid & 1) * 16;
    unsigned long long ga =
        (unsigned long long)(uintptr_t)(vb + (size_t)row * kQKVC + half);
    unsigned lds_off = (unsigned)(uintptr_t)(&v_lds[row * 32 + half]);
    asm volatile("global_load_async_to_lds_b128 %0, %1, off"
                 :: "v"(lds_off), "v"(ga) : "memory");
    asm volatile("global_load_async_to_lds_b128 %0, %1, off offset:16"
                 :: "v"(lds_off), "v"(ga) : "memory");
    asm volatile("s_wait_asynccnt 0" ::: "memory");
  }
  __syncthreads();

  int m0 = wv << 4;
  // ---- scores: S = Q K^T (one WMMA K-step, head dim = 32) ----
  v16h aq = ldfrag(qb + (size_t)(m0 + lr) * kQKVC, hi);
  v8f s0 = {}, s1 = {}, s2 = {}, s3 = {};
  {
    v16h b0 = ldfrag(kb + (size_t)(lr) * kQKVC, hi);
    v16h b1 = ldfrag(kb + (size_t)(16 + lr) * kQKVC, hi);
    v16h b2 = ldfrag(kb + (size_t)(32 + lr) * kQKVC, hi);
    v16h b3 = ldfrag(kb + (size_t)(48 + lr) * kQKVC, hi);
    s0 = wmma_f16(aq, b0, s0);
    s1 = wmma_f16(aq, b1, s1);
    s2 = wmma_f16(aq, b2, s2);
    s3 = wmma_f16(aq, b3, s3);
  }
  const float scale = 0.17677669529663687f;  // 32^-0.5
  const float* bf = bias_full + (size_t)head * (kNP * kNP);
#pragma unroll
  for (int i = 0; i < 8; ++i) {
    int row = m0 + (hi << 3) + i;
    const float* br = bf + row * kNP + lr;
    s0[i] = s0[i] * scale + br[0];
    s1[i] = s1[i] * scale + br[16];
    s2[i] = s2[i] * scale + br[32];
    s3[i] = s3[i] * scale + br[48];
  }
  // ---- softmax over 64 cols: rows live in 16-lane half groups ----
  float inv[8];
#pragma unroll
  for (int i = 0; i < 8; ++i) {
    float m = fmaxf(fmaxf(s0[i], s1[i]), fmaxf(s2[i], s3[i]));
    for (int off = 1; off < 16; off <<= 1) m = fmaxf(m, __shfl_xor(m, off, 32));
    s0[i] = __expf(s0[i] - m);
    s1[i] = __expf(s1[i] - m);
    s2[i] = __expf(s2[i] - m);
    s3[i] = __expf(s3[i] - m);
    float su = s0[i] + s1[i] + s2[i] + s3[i];
    for (int off = 1; off < 16; off <<= 1) su += __shfl_xor(su, off, 32);
    inv[i] = 1.0f / su;
  }
  // stage P (f16) in LDS: C-layout -> A-layout transpose (same-wave rows only)
#pragma unroll
  for (int i = 0; i < 8; ++i) {
    int row = m0 + (hi << 3) + i;
    p_lds[row * 64 + lr]      = (_Float16)(s0[i] * inv[i]);
    p_lds[row * 64 + 16 + lr] = (_Float16)(s1[i] * inv[i]);
    p_lds[row * 64 + 32 + lr] = (_Float16)(s2[i] * inv[i]);
    p_lds[row * 64 + 48 + lr] = (_Float16)(s3[i] * inv[i]);
  }
  // ---- O = P V (K = 64, two WMMA steps; V as B operand from LDS) ----
  v8f o0 = {}, o1 = {};
#pragma unroll
  for (int ks = 0; ks < 2; ++ks) {
    const _Float16* prow = &p_lds[(m0 + lr) * 64 + ks * 32 + hi * 8];
    v16h a;
#pragma unroll
    for (int e = 0; e < 8; ++e) a[e] = prow[e];
#pragma unroll
    for (int e = 0; e < 8; ++e) a[8 + e] = prow[16 + e];
    int kk = ks * 32 + hi * 8;
    v16h b0, b1;
#pragma unroll
    for (int e = 0; e < 8; ++e) {
      b0[e]     = v_lds[(kk + e) * 32 + lr];
      b0[8 + e] = v_lds[(kk + 16 + e) * 32 + lr];
      b1[e]     = v_lds[(kk + e) * 32 + 16 + lr];
      b1[8 + e] = v_lds[(kk + 16 + e) * 32 + 16 + lr];
    }
    o0 = wmma_f16(a, b0, o0);
    o1 = wmma_f16(a, b1, o1);
  }
  // store O[16x32] into attnout16 [NW*64][C] at cols head*32..
#pragma unroll
  for (int i = 0; i < 8; ++i) {
    size_t row = (size_t)w * kNP + m0 + (hi << 3) + i;
    outp[row * kC + head * 32 + lr]      = (_Float16)o0[i];
    outp[row * kC + head * 32 + 16 + lr] = (_Float16)o1[i];
  }
}

// ---------------------------------------------------------------------------
// Un-window + residual: x1[tok][c] = x[tok][c] + proj16[windowrow][c]
// ---------------------------------------------------------------------------
__global__ __launch_bounds__(256) void unwin_res_kernel(
    const float* __restrict__ x, const _Float16* __restrict__ proj,
    float* __restrict__ x1) {
  size_t idx = (size_t)blockIdx.x * 256 + threadIdx.x;
  if (idx >= (size_t)kTOK * kC) return;
  int c = (int)(idx % kC);
  size_t tok = idx / kC;
  int b = (int)(tok / kHW);
  int hw = (int)(tok - (size_t)b * kHW);
  int hrow = hw / 28, wcol = hw - hrow * 28;
  int wh = hrow / 7, r = hrow - wh * 7;
  int ww = wcol / 7, cc = wcol - ww * 7;
  int wdx = (b << 4) + (wh << 2) + ww;
  size_t wrow = (size_t)wdx * kNP + (r * 7 + cc);
  x1[idx] = x[idx] + (float)proj[wrow * kC + c];
}

// ---------------------------------------------------------------------------
// Depthwise 3x3 conv + BN (token-major f32 in/out, zero padding)
// ---------------------------------------------------------------------------
__global__ __launch_bounds__(256) void dwconv_bn_kernel(
    const float* __restrict__ x1, const float* __restrict__ cw,
    const float* __restrict__ bg, const float* __restrict__ bb,
    const float* __restrict__ bm, const float* __restrict__ bv,
    float* __restrict__ x2) {
  size_t idx = (size_t)blockIdx.x * 256 + threadIdx.x;
  if (idx >= (size_t)kTOK * kC) return;
  int c = (int)(idx % kC);
  size_t tok = idx / kC;
  int b = (int)(tok / kHW);
  int hw = (int)(tok - (size_t)b * kHW);
  int h = hw / 28, wd = hw - h * 28;
  const float* wc = cw + c * 9;
  const float* xb = x1 + (size_t)b * kHW * kC + c;
  float s = 0.f;
#pragma unroll
  for (int ky = 0; ky < 3; ++ky) {
#pragma unroll
    for (int kx = 0; kx < 3; ++kx) {
      int py = h + ky - 1, px = wd + kx - 1;
      if (py >= 0 && py < 28 && px >= 0 && px < 28)
        s += wc[ky * 3 + kx] * xb[(size_t)(py * 28 + px) * kC];
    }
  }
  float inv = rsqrtf(bv[c] + 1e-5f) * bg[c];
  x2[idx] = (s - bm[c]) * inv + bb[c];
}

__global__ __launch_bounds__(256) void f32_to_f16_kernel(
    const float* __restrict__ src, _Float16* __restrict__ dst, int n) {
  int i = blockIdx.x * 256 + threadIdx.x;
  if (i < n) dst[i] = (_Float16)src[i];
}

// ---------------------------------------------------------------------------
extern "C" void kernel_launch(void* const* d_in, const int* in_sizes, int n_in,
                              void* d_out, int out_size, void* d_ws,
                              size_t ws_size, hipStream_t stream) {
  const float* x        = (const float*)d_in[0];
  const float* gamma_a  = (const float*)d_in[1];
  const float* beta_a   = (const float*)d_in[2];
  const float* qkv_w    = (const float*)d_in[3];
  const float* qkv_b    = (const float*)d_in[4];
  const float* proj_w   = (const float*)d_in[5];
  const float* proj_b   = (const float*)d_in[6];
  const float* attn_bs  = (const float*)d_in[7];
  const float* conv_w   = (const float*)d_in[8];
  const float* bn_gamma = (const float*)d_in[9];
  const float* bn_beta  = (const float*)d_in[10];
  const float* bn_mean  = (const float*)d_in[11];
  const float* bn_var   = (const float*)d_in[12];
  const float* gamma_m  = (const float*)d_in[13];
  const float* beta_m   = (const float*)d_in[14];
  const float* fc1_w    = (const float*)d_in[15];
  const float* fc1_b    = (const float*)d_in[16];
  const float* fc2_w    = (const float*)d_in[17];
  const float* fc2_b    = (const float*)d_in[18];
  const int*   bidx     = (const int*)d_in[19];
  float* out = (float*)d_out;
  int n_off = in_sizes[7] / kNH;

  char* ws = (char*)d_ws;
  size_t off = 0;
  auto alloc = [&](size_t bytes) -> void* {
    void* p = ws + off;
    off += (bytes + 255) & ~(size_t)255;
    return p;
  };
  size_t winrows = (size_t)kNW * kNP;  // 65536 padded window rows
  // h16: LN output / reused for proj-out and MLP-LN output
  _Float16* h16 = (_Float16*)alloc(winrows * kC * 2);
  // big: qkv16 (65536x1728) then reused for MLP hidden (50176x2304)
  size_t bigA = winrows * kQKVC * 2;
  size_t bigB = (size_t)kTOK * kHID * 2;
  _Float16* big = (_Float16*)alloc(bigA > bigB ? bigA : bigB);
  _Float16* ao16 = (_Float16*)alloc(winrows * kC * 2);          // attn out
  float* x1 = (float*)alloc((size_t)kTOK * kC * 4);             // post-attn res
  float* x2 = (float*)alloc((size_t)kTOK * kC * 4);             // post-conv/BN
  _Float16* qkvw16  = (_Float16*)alloc((size_t)kQKVC * kC * 2);
  _Float16* projw16 = (_Float16*)alloc((size_t)kC * kC * 2);
  _Float16* fc1w16  = (_Float16*)alloc((size_t)kHID * kC * 2);
  _Float16* fc2w16  = (_Float16*)alloc((size_t)kC * kHID * 2);
  float* bias_full  = (float*)alloc((size_t)kNH * kNP * kNP * 4);

  auto cvt = [&](const float* s, _Float16* d, int n) {
    f32_to_f16_kernel<<<(n + 255) / 256, 256, 0, stream>>>(s, d, n);
  };
  cvt(qkv_w, qkvw16, kQKVC * kC);
  cvt(proj_w, projw16, kC * kC);
  cvt(fc1_w, fc1w16, kHID * kC);
  cvt(fc2_w, fc2w16, kC * kHID);
  expand_bias_kernel<<<(kNH * kNP * kNP + 255) / 256, 256, 0, stream>>>(
      attn_bs, bidx, bias_full, n_off);

  // 1) windowed LN -> h16 (padded)
  ln_win_kernel<<<kNW * kNP, 256, 0, stream>>>(x, gamma_a, beta_a, h16);
  // 2) qkv GEMM: [65536,576] x [1728,576]^T -> big
  gemm_wmma<EPI_F16><<<(65536 / 32) * (kQKVC / 64) / 8, 256, 0, stream>>>(
      h16, qkvw16, qkv_b, big, nullptr, 65536, kC, kQKVC);
  // 3) windowed attention
  attn_kernel<<<kNW * kNH, 128, 0, stream>>>(big, bias_full, ao16);
  // 4) proj GEMM: [65536,576] x [576,576]^T -> h16 (reuse)
  gemm_wmma<EPI_F16><<<(65536 / 32) * (kC / 64) / 8, 256, 0, stream>>>(
      ao16, projw16, proj_b, h16, nullptr, 65536, kC, kC);
  // 5) un-window + residual -> x1
  unwin_res_kernel<<<(int)(((size_t)kTOK * kC + 255) / 256), 256, 0, stream>>>(
      x, h16, x1);
  // 6) depthwise conv + BN -> x2
  dwconv_bn_kernel<<<(int)(((size_t)kTOK * kC + 255) / 256), 256, 0, stream>>>(
      x1, conv_w, bn_gamma, bn_beta, bn_mean, bn_var, x2);
  // 7) token LN -> h16 (reuse)
  ln_tok_kernel<<<kTOK, 256, 0, stream>>>(x2, gamma_m, beta_m, h16);
  // 8) fc1 GEMM + exact GELU: [50176,576] x [2304,576]^T -> big (reuse)
  gemm_wmma<EPI_GELU><<<(kTOK / 32) * (kHID / 64) / 8, 256, 0, stream>>>(
      h16, fc1w16, fc1_b, big, nullptr, kTOK, kC, kHID);
  // 9) fc2 GEMM + residual(x2) -> d_out (f32)
  gemm_wmma<EPI_RESF32><<<(kTOK / 32) * (kC / 64) / 8, 256, 0, stream>>>(
      big, fc2w16, fc2_b, out, x2, kTOK, kHID, kC);
}
